// Network_74483322847706
// MI455X (gfx1250) — matrix operand order repaired and checked by reading, not compiled, not run on hardware
//
#include <hip/hip_runtime.h>
#include <hip/hip_bf16.h>
#include <math.h>
#include <cstdint>

// ---------------- types ----------------
typedef __attribute__((ext_vector_type(16))) _Float16 v16h;
typedef __attribute__((ext_vector_type(8)))  _Float16 v8h;
typedef __attribute__((ext_vector_type(4)))  _Float16 v4h;
typedef __attribute__((ext_vector_type(8)))  float    v8f;
typedef __attribute__((ext_vector_type(4)))  unsigned int u32x4;
typedef __attribute__((ext_vector_type(8)))  int          i32x8;
typedef __attribute__((ext_vector_type(4)))  int          i32x4;

union H16u { v16h v; v8h h[2]; _Float16 e[16]; };

#define N_RAYS   1024
#define NS_C     64
#define NS_F     128
#define STRIDE_H 320   // activation row stride (f16 elems); >= max K (320)

#if defined(__HIP_DEVICE_COMPILE__) && __has_builtin(__builtin_amdgcn_tensor_load_to_lds)
#define USE_TDM 1
#else
#define USE_TDM 0
#endif

// ---------------- weight prep: f32 [din x dout] row-major -> f16 transposed [dout x KP]
// with optional zero-row insertion (for the skip-concat padding gap) and tail zero pad.
__global__ void prep_weights(const float* __restrict__ Wsrc, int din, int dout,
                             _Float16* __restrict__ Bt, int KP, int insertAt)
{
    int idx = blockIdx.x * blockDim.x + threadIdx.x;
    if (idx >= dout * KP) return;
    int c = idx / KP;
    int k = idx % KP;
    int src = k;
    if (insertAt >= 0) {
        if (k == insertAt)      src = -1;
        else if (k > insertAt)  src = k - 1;
    }
    float v = 0.0f;
    if (src >= 0 && src < din) v = Wsrc[(size_t)src * dout + c];
    Bt[(size_t)c * KP + k] = (_Float16)v;
}

// ---------------- per-ray setup: coarse z_vals, view-dir posenc (27 -> pad 32), |d|
__global__ void ray_setup(const float* __restrict__ rb, float* __restrict__ zc,
                          _Float16* __restrict__ de, float* __restrict__ dnorm)
{
    int r = blockIdx.x * blockDim.x + threadIdx.x;
    if (r >= N_RAYS) return;
    const float* row = rb + r * 11;
    float nearv = row[6], farv = row[7];
    for (int s = 0; s < NS_C; ++s) {
        float t = (float)s / (float)(NS_C - 1);
        zc[r * NS_C + s] = nearv * (1.0f - t) + farv * t;
    }
    float dx = row[3], dy = row[4], dz = row[5];
    dnorm[r] = sqrtf(dx * dx + dy * dy + dz * dz);
    float v[3] = { row[8], row[9], row[10] };
    union { _Float16 e[32]; v8h v[4]; } buf;
    buf.e[0] = (_Float16)v[0]; buf.e[1] = (_Float16)v[1]; buf.e[2] = (_Float16)v[2];
#pragma unroll
    for (int l = 0; l < 4; ++l) {
        float f = (float)(1 << l);
#pragma unroll
        for (int i = 0; i < 3; ++i) {
            buf.e[3 + l * 6 + i]     = (_Float16)__sinf(f * v[i]);
            buf.e[3 + l * 6 + 3 + i] = (_Float16)__cosf(f * v[i]);
        }
    }
#pragma unroll
    for (int j = 27; j < 32; ++j) buf.e[j] = (_Float16)0.0f;
    v8h* dst = (v8h*)(de + r * 32);
#pragma unroll
    for (int i = 0; i < 4; ++i) dst[i] = buf.v[i];
}

// ---------------- point posenc: pts = o + d*z; 63-ch encoding padded to 64, f16
__global__ void posenc_pts(const float* __restrict__ rb, const float* __restrict__ z,
                           int nsamp, _Float16* __restrict__ X, int M)
{
    int m = blockIdx.x * blockDim.x + threadIdx.x;
    if (m >= M) return;
    int r = m / nsamp, s = m % nsamp;
    const float* row = rb + r * 11;
    float zv = z[r * nsamp + s];
    float p[3];
#pragma unroll
    for (int i = 0; i < 3; ++i) p[i] = row[i] + row[3 + i] * zv;
    union { _Float16 e[64]; v8h v[8]; } buf;
    buf.e[0] = (_Float16)p[0]; buf.e[1] = (_Float16)p[1]; buf.e[2] = (_Float16)p[2];
#pragma unroll
    for (int l = 0; l < 10; ++l) {
        float f = (float)(1 << l);
#pragma unroll
        for (int i = 0; i < 3; ++i) {
            buf.e[3 + l * 6 + i]     = (_Float16)__sinf(f * p[i]);
            buf.e[3 + l * 6 + 3 + i] = (_Float16)__cosf(f * p[i]);
        }
    }
    buf.e[63] = (_Float16)0.0f;
    v8h* dst = (v8h*)(X + (size_t)m * 64);
#pragma unroll
    for (int i = 0; i < 8; ++i) dst[i] = buf.v[i];
}

// ---------------- WMMA GEMM: C[M x N] = act(A[M x KP] * B[KP x N] + bias)
// A: f16 row-major (global). Bt: pre-transposed weights, Bt[c*KP + k] (global).
// Per block: TDM stages the 64-column weight slab (KP x 64 f16) into LDS once;
// waves then feed v_wmma_f32_16x16x32_f16 with BOTH A (global) and B (LDS)
// double-buffered in registers, so each K-step's waits hide behind the previous
// step's 4 WMMAs. Block = 8 wave32s -> tile 128 rows x 64 cols.
// M must be a multiple of 128, KP of 32.
__global__ __launch_bounds__(256) void mlp_gemm(
    const _Float16* __restrict__ A, int lda,
    const _Float16* __restrict__ Bt, int KP,
    const float* __restrict__ bias,
    _Float16* __restrict__ C, int ldc, int coff,
    int M, int relu)
{
    __shared__ _Float16 ldsB[320 * 64];   // 40 KB: full K for 64 output columns

    const int wave = threadIdx.x >> 5;
    const int lane = threadIdx.x & 31;
    const int l16  = lane & 15;
    const int lhi  = lane >> 4;
    const int row0 = blockIdx.x * 128 + wave * 16;
    const int col0 = blockIdx.y * 64;

    // ---- stage weight slab into LDS ----
#if USE_TDM
    if (wave == 0) {
        unsigned long long ga = (unsigned long long)(uintptr_t)(Bt + (size_t)col0 * KP);
        unsigned lds_off = (unsigned)(uintptr_t)ldsB;   // low 32 bits = LDS byte address
        u32x4 g0;
        g0.x = 1u;                                        // count=1, user mode
        g0.y = lds_off;                                   // lds_addr
        g0.z = (unsigned)(ga & 0xFFFFFFFFu);              // global_addr[31:0]
        g0.w = (unsigned)((ga >> 32) & 0x01FFFFFFu) | (2u << 30);  // global_addr[56:32], type=2
        i32x8 g1;
        g1[0] = (1 << 16);                                // data_size=1 (2 bytes)
        g1[1] = (int)((KP & 0xFFFF) << 16);               // tensor_dim0[15:0]
        g1[2] = (int)(((unsigned)KP >> 16) | (64u << 16));// tensor_dim0[31:16] | tensor_dim1[15:0]=64
        g1[3] = (int)((unsigned)KP << 16);                // tensor_dim1[31:16]=0 | tile_dim0=KP
        g1[4] = 64;                                       // tile_dim1=64, tile_dim2=0
        g1[5] = KP;                                       // tensor_dim0_stride[31:0]
        g1[6] = 0;
        g1[7] = 0;
        i32x4 gz; gz[0] = 0; gz[1] = 0; gz[2] = 0; gz[3] = 0;
#if __clang_major__ >= 23
        i32x8 gz8; gz8[0]=0; gz8[1]=0; gz8[2]=0; gz8[3]=0; gz8[4]=0; gz8[5]=0; gz8[6]=0; gz8[7]=0;
        __builtin_amdgcn_tensor_load_to_lds(g0, g1, gz, gz, gz8, 0);
#else
        __builtin_amdgcn_tensor_load_to_lds(g0, g1, gz, gz, 0);
#endif
        __builtin_amdgcn_s_wait_tensorcnt(0);
    }
    __syncthreads();
#else
    {
        const v8h* src = (const v8h*)(Bt + (size_t)col0 * KP);
        v8h* dst = (v8h*)ldsB;
        int nchunk = (KP * 64) >> 3;
        for (int i = threadIdx.x; i < nchunk; i += 256) dst[i] = src[i];
    }
    __syncthreads();
#endif

    v8f acc[4] = {};

    // A fragment base: ISA 16-bit A layout -> lane holds two contiguous 16B K-chunks
    const _Float16* aptr = A + (size_t)(row0 + l16) * lda + 8 * lhi;
    // B fragment base in LDS: lane = output column, two 16B K-chunks (same layout as A)
    const _Float16* bbase = ldsB + l16 * KP + 8 * lhi;

    struct BFr { H16u b[4]; };
    auto loadA = [&](H16u& af, int k0) {
        af.h[0] = *(const v8h*)(aptr + k0);
        af.h[1] = *(const v8h*)(aptr + k0 + 16);
    };
    auto loadB = [&](BFr& bf, int k0) {
#pragma unroll
        for (int cb = 0; cb < 4; ++cb) {
            const v8h* bp = (const v8h*)(bbase + cb * 16 * KP + k0);
            bf.b[cb].h[0] = bp[0];
            bf.b[cb].h[1] = bp[2];
        }
    };

    H16u a;  BFr bc;
    loadA(a, 0);
    loadB(bc, 0);
    int k0 = 0;
    for (; k0 + 32 < KP; k0 += 32) {
        H16u an; BFr bn;                 // prefetch next K-step (A from global, B from LDS)
        loadA(an, k0 + 32);
        loadB(bn, k0 + 32);
#pragma unroll
        for (int cb = 0; cb < 4; ++cb)
            acc[cb] = __builtin_amdgcn_wmma_f32_16x16x32_f16(
                false, a.v, false, bc.b[cb].v, (short)0, acc[cb], false, false);
        a = an; bc = bn;
    }
#pragma unroll
    for (int cb = 0; cb < 4; ++cb)       // epilogue
        acc[cb] = __builtin_amdgcn_wmma_f32_16x16x32_f16(
            false, a.v, false, bc.b[cb].v, (short)0, acc[cb], false, false);

    // C layout: lane -> col l16, rows lhi*8 + i (i = VGPR index)
#pragma unroll
    for (int cb = 0; cb < 4; ++cb) {
        int c = col0 + cb * 16 + l16;
        float bv = bias[c];
        union { v8f v; float f[8]; } u; u.v = acc[cb];
#pragma unroll
        for (int i = 0; i < 8; ++i) {
            float val = u.f[i] + bv;
            if (relu) val = fmaxf(val, 0.0f);
            int rr = row0 + lhi * 8 + i;
            C[(size_t)rr * ldc + coff + c] = (_Float16)val;
        }
    }
}

// ---------------- assemble skip-concat: pe (padded 64) into cols 0..63 of H
__global__ void copy_X_to_H(const _Float16* __restrict__ X, _Float16* __restrict__ H, int M)
{
    int idx = blockIdx.x * blockDim.x + threadIdx.x;
    if (idx >= M * 64) return;
    int m = idx >> 6, j = idx & 63;
    H[(size_t)m * STRIDE_H + j] = X[(size_t)m * 64 + j];
}

// ---------------- broadcast view-dir enc (32 padded) into cols 256..287 of H
__global__ void bcast_de(const _Float16* __restrict__ de, _Float16* __restrict__ H,
                         int M, int nsamp)
{
    int idx = blockIdx.x * blockDim.x + threadIdx.x;
    if (idx >= M * 32) return;
    int m = idx >> 5, j = idx & 31;
    H[(size_t)m * STRIDE_H + 256 + j] = de[(m / nsamp) * 32 + j];
}

// ---------------- alpha head: 256 -> 1; one wave32 per sample row
__global__ __launch_bounds__(256) void head_alpha(const _Float16* __restrict__ H,
                                                  const float* __restrict__ wa,
                                                  const float* __restrict__ ba,
                                                  float* __restrict__ raw, int M)
{
    int gw   = (blockIdx.x * 256 + threadIdx.x) >> 5;
    int lane = threadIdx.x & 31;
    if (gw >= M) return;
    v8h hv = *(const v8h*)(H + (size_t)gw * STRIDE_H + lane * 8);
    const float* wp = wa + lane * 8;
    float s = 0.0f;
#pragma unroll
    for (int i = 0; i < 8; ++i) s += (float)hv[i] * wp[i];
#pragma unroll
    for (int off = 16; off > 0; off >>= 1) s += __shfl_down(s, off, 32);
    if (lane == 0) raw[gw * 4 + 3] = s + ba[0];
}

// ---------------- rgb head: 128 -> 3; one wave32 per sample row
__global__ __launch_bounds__(256) void head_rgb(const _Float16* __restrict__ H,
                                                const float* __restrict__ wr,
                                                const float* __restrict__ br,
                                                float* __restrict__ raw, int M)
{
    int gw   = (blockIdx.x * 256 + threadIdx.x) >> 5;
    int lane = threadIdx.x & 31;
    if (gw >= M) return;
    v4h hv = *(const v4h*)(H + (size_t)gw * STRIDE_H + lane * 4);
    float s0 = 0.f, s1 = 0.f, s2 = 0.f;
#pragma unroll
    for (int i = 0; i < 4; ++i) {
        float h = (float)hv[i];
        const float* w = wr + (lane * 4 + i) * 3;
        s0 += h * w[0]; s1 += h * w[1]; s2 += h * w[2];
    }
#pragma unroll
    for (int off = 16; off > 0; off >>= 1) {
        s0 += __shfl_down(s0, off, 32);
        s1 += __shfl_down(s1, off, 32);
        s2 += __shfl_down(s2, off, 32);
    }
    if (lane == 0) {
        raw[gw * 4 + 0] = s0 + br[0];
        raw[gw * 4 + 1] = s1 + br[1];
        raw[gw * 4 + 2] = s2 + br[2];
    }
}

// ---------------- volume rendering (raw2outputs); one thread per ray
__global__ void composite(const float* __restrict__ raw, const float* __restrict__ z,
                          const float* __restrict__ dnorm, int nsamp,
                          float* __restrict__ out_rgb, float* __restrict__ out_disp,
                          float* __restrict__ out_acc, float* __restrict__ weights_out)
{
    int r = blockIdx.x * blockDim.x + threadIdx.x;
    if (r >= N_RAYS) return;
    float T = 1.0f, rm0 = 0.f, rm1 = 0.f, rm2 = 0.f, depth = 0.f, acc = 0.f;
    float dn = dnorm[r];
    for (int s = 0; s < nsamp; ++s) {
        float zv = z[r * nsamp + s];
        float dist = ((s < nsamp - 1) ? (z[r * nsamp + s + 1] - zv) : 1e10f) * dn;
        int base = (r * nsamp + s) * 4;
        float a = 1.0f - __expf(-fmaxf(raw[base + 3], 0.0f) * dist);
        float w = a * T;
        T *= (1.0f - a + 1e-10f);
        rm0 += w * (1.0f / (1.0f + __expf(-raw[base + 0])));
        rm1 += w * (1.0f / (1.0f + __expf(-raw[base + 1])));
        rm2 += w * (1.0f / (1.0f + __expf(-raw[base + 2])));
        depth += w * zv;
        acc   += w;
        if (weights_out) weights_out[r * nsamp + s] = w;
    }
    out_rgb[r * 3 + 0] = rm0;
    out_rgb[r * 3 + 1] = rm1;
    out_rgb[r * 3 + 2] = rm2;
    out_disp[r] = 1.0f / fmaxf(1e-10f, depth / fmaxf(acc, 1e-10f));
    out_acc[r]  = acc;
}

// ---------------- sample_pdf + sorted merge + z_std; one thread per ray
__global__ void pdf_and_merge(const float* __restrict__ zc, const float* __restrict__ wts,
                              float* __restrict__ zall, float* __restrict__ zstd)
{
    int r = blockIdx.x * blockDim.x + threadIdx.x;
    if (r >= N_RAYS) return;
    const float* z = zc + r * NS_C;
    float bins[63], cdf[63], pdf[62], zs[64];
    for (int i = 0; i < 63; ++i) bins[i] = 0.5f * (z[i] + z[i + 1]);
    float sum = 0.0f;
    for (int i = 0; i < 62; ++i) { pdf[i] = wts[r * NS_C + 1 + i] + 1e-5f; sum += pdf[i]; }
    float inv = 1.0f / sum;
    cdf[0] = 0.0f;
    for (int i = 1; i < 63; ++i) cdf[i] = cdf[i - 1] + pdf[i - 1] * inv;
    float mean = 0.0f;
    for (int j = 0; j < 64; ++j) {
        float u = (float)j / 63.0f;
        int ind = 0;
        for (int i = 0; i < 63; ++i) ind += (cdf[i] <= u) ? 1 : 0;   // searchsorted right
        int below = ind - 1; if (below < 0) below = 0;
        int above = ind;     if (above > 62) above = 62;
        float d = cdf[above] - cdf[below];
        if (d < 1e-5f) d = 1.0f;
        float t = (u - cdf[below]) / d;
        float s = bins[below] + t * (bins[above] - bins[below]);
        zs[j] = s; mean += s;
    }
    mean *= (1.0f / 64.0f);
    float var = 0.0f;
    for (int j = 0; j < 64; ++j) { float dd = zs[j] - mean; var += dd * dd; }
    zstd[r] = sqrtf(var * (1.0f / 64.0f));
    // merge two sorted 64-lists
    int i = 0, j = 0;
    float* za = zall + r * NS_F;
    for (int k = 0; k < NS_F; ++k) {
        float a = (i < 64) ? z[i]  : 3.4e38f;
        float b = (j < 64) ? zs[j] : 3.4e38f;
        if (a <= b) { za[k] = a; ++i; } else { za[k] = b; ++j; }
    }
}

// ---------------- host orchestration ----------------
extern "C" void kernel_launch(void* const* d_in, const int* in_sizes, int n_in,
                              void* d_out, int out_size, void* d_ws, size_t ws_size,
                              hipStream_t stream)
{
    (void)in_sizes; (void)n_in; (void)out_size; (void)ws_size;
    const float* rb = (const float*)d_in[0];
    float* out = (float*)d_out;

    // carve workspace
    char* ws = (char*)d_ws;
    size_t off = 0;
    auto carve = [&](size_t bytes) -> char* {
        char* p = ws + off;
        off += (bytes + 255) & ~(size_t)255;
        return p;
    };
    const int MMAX = N_RAYS * NS_F;
    _Float16* X   = (_Float16*)carve((size_t)MMAX * 64 * 2);
    _Float16* HA  = (_Float16*)carve((size_t)MMAX * STRIDE_H * 2);
    _Float16* HB  = (_Float16*)carve((size_t)MMAX * STRIDE_H * 2);
    float*    raw = (float*)   carve((size_t)MMAX * 4 * 4);
    float*    zc  = (float*)   carve((size_t)N_RAYS * NS_C * 4);
    float*    zal = (float*)   carve((size_t)N_RAYS * NS_F * 4);
    float*    wts = (float*)   carve((size_t)N_RAYS * NS_C * 4);
    _Float16* de  = (_Float16*)carve((size_t)N_RAYS * 32 * 2);
    float*    dnm = (float*)   carve((size_t)N_RAYS * 4);
    _Float16* bt[10];
    for (int i = 0; i < 10; ++i) bt[i] = (_Float16*)carve((size_t)320 * 256 * 2);

    ray_setup<<<(N_RAYS + 255) / 256, 256, 0, stream>>>(rb, zc, de, dnm);

    auto run_net = [&](int pbase, int M, int nsamp, const float* zbuf,
                       float* orgb, float* odisp, float* oacc, float* wout) {
        auto P = [&](int i) -> const float* { return (const float*)d_in[pbase + i]; };
        struct LayerDesc { int pidx, din, dout, KP, ins; };
        const LayerDesc L[10] = {
            { 0,  63, 256,  64, -1},   // w0 (posenc 63 -> pad 64)
            { 2, 256, 256, 256, -1},   // w1
            { 4, 256, 256, 256, -1},   // w2
            { 6, 256, 256, 256, -1},   // w3
            { 8, 256, 256, 256, -1},   // w4
            {10, 319, 256, 320, 63},   // w5 (skip concat: [pe63, pad1, h256])
            {12, 256, 256, 256, -1},   // w6
            {14, 256, 256, 256, -1},   // w7
            {16, 256, 256, 256, -1},   // wf
            {20, 283, 128, 288, -1},   // wv ([feat256, de27] -> pad 288)
        };
        for (int i = 0; i < 10; ++i) {
            int n = L[i].dout * L[i].KP;
            prep_weights<<<(n + 255) / 256, 256, 0, stream>>>(
                P(L[i].pidx), L[i].din, L[i].dout, bt[i], L[i].KP, L[i].ins);
        }
        posenc_pts<<<(M + 255) / 256, 256, 0, stream>>>(rb, zbuf, nsamp, X, M);

        dim3 blk(256);
        dim3 g4(M / 128, 4), g2(M / 128, 2);
        mlp_gemm<<<g4, blk, 0, stream>>>(X,  64,       bt[0],  64, P(1),  HA, STRIDE_H,  0, M, 1);
        mlp_gemm<<<g4, blk, 0, stream>>>(HA, STRIDE_H, bt[1], 256, P(3),  HB, STRIDE_H,  0, M, 1);
        mlp_gemm<<<g4, blk, 0, stream>>>(HB, STRIDE_H, bt[2], 256, P(5),  HA, STRIDE_H,  0, M, 1);
        mlp_gemm<<<g4, blk, 0, stream>>>(HA, STRIDE_H, bt[3], 256, P(7),  HB, STRIDE_H,  0, M, 1);
        mlp_gemm<<<g4, blk, 0, stream>>>(HB, STRIDE_H, bt[4], 256, P(9),  HA, STRIDE_H, 64, M, 1);
        copy_X_to_H<<<(M * 64 + 255) / 256, 256, 0, stream>>>(X, HA, M);
        mlp_gemm<<<g4, blk, 0, stream>>>(HA, STRIDE_H, bt[5], 320, P(11), HB, STRIDE_H,  0, M, 1);
        mlp_gemm<<<g4, blk, 0, stream>>>(HB, STRIDE_H, bt[6], 256, P(13), HA, STRIDE_H,  0, M, 1);
        mlp_gemm<<<g4, blk, 0, stream>>>(HA, STRIDE_H, bt[7], 256, P(15), HB, STRIDE_H,  0, M, 1);
        // heads: alpha from final h (HB), feature (no relu) -> HA, concat de, view MLP, rgb
        head_alpha<<<M / 8, blk, 0, stream>>>(HB, P(18), P(19), raw, M);
        mlp_gemm<<<g4, blk, 0, stream>>>(HB, STRIDE_H, bt[8], 256, P(17), HA, STRIDE_H,  0, M, 0);
        bcast_de<<<(M * 32 + 255) / 256, 256, 0, stream>>>(de, HA, M, nsamp);
        mlp_gemm<<<g2, blk, 0, stream>>>(HA, STRIDE_H, bt[9], 288, P(21), HB, STRIDE_H,  0, M, 1);
        head_rgb<<<M / 8, blk, 0, stream>>>(HB, P(22), P(23), raw, M);
        composite<<<(N_RAYS + 255) / 256, 256, 0, stream>>>(raw, zbuf, dnm, nsamp,
                                                            orgb, odisp, oacc, wout);
    };

    // out layout (floats): rgb_map[3072] | disp[1024] | acc[1024] | rgb0[3072] | disp0[1024] | acc0[1024] | z_std[1024]
    run_net(1, N_RAYS * NS_C, NS_C, zc, out + 5120, out + 8192, out + 9216, wts);   // coarse
    pdf_and_merge<<<(N_RAYS + 255) / 256, 256, 0, stream>>>(zc, wts, zal, out + 10240);
    run_net(25, N_RAYS * NS_F, NS_F, zal, out + 0, out + 3072, out + 4096, nullptr); // fine
}
